// GraphSAGE_48258252538107
// MI455X (gfx1250) — matrix operand order repaired and checked by reading, not compiled, not run on hardware
//
#include <hip/hip_runtime.h>

// GraphSAGE (3-layer, mean aggregation) for MI455X / gfx1250.
//
// Dominant cost: 3x (mask^T[16384x16384] @ x[16384x256]) = 412 GFLOP, plus
// 1 GiB fp32 adjacency traffic. Strategy:
//   - adj read ONCE: transposed+converted to bf16 maskT [dst][src] (A-ready),
//     degree counting fused into the same pass.
//   - activations kept feature-major xT [feat][node] (B-ready), so the big
//     aggregation GEMM stages both operands as contiguous 16B chunks via
//     global_load_async_to_lds_b128 (double-buffered, ASYNCcnt+barrier),
//     feeding v_wmma_f32_16x16x32_bf16.
//   - WMMA C-layout makes transposed epilogues single b128 stores per tile.
// Fallback path (small workspace): round-1 on-the-fly conversion kernels.

#define NN 16384   // nodes
#define DF 256     // feature width (IN == HID == 256)

typedef __bf16 bf16x16 __attribute__((ext_vector_type(16)));
typedef float  f32x8   __attribute__((ext_vector_type(8)));

union Frag { uint4 u[2]; bf16x16 v; };

__device__ __forceinline__ unsigned short f2bf(float f) {
  unsigned int u = __float_as_uint(f);
  u += 0x7FFFu + ((u >> 16) & 1u);           // round-to-nearest-even
  return (unsigned short)(u >> 16);
}

// LDS byte offset of a __shared__ object: per ISA, generic LDS address low
// 32 bits are the LDS offset (aperture lives in addr[63:32]).
__device__ __forceinline__ unsigned ldsOff(const void* p) {
  return (unsigned)(unsigned long long)p;
}

// Async 16B global->LDS copy (CDNA5), tracked by ASYNCcnt.
__device__ __forceinline__ void asyncCopy16(const void* lds, const void* gptr) {
  asm volatile("global_load_async_to_lds_b128 %0, %1, off"
               :: "v"(ldsOff(lds)), "v"(gptr) : "memory");
}
__device__ __forceinline__ void waitAsync0() {
  asm volatile("s_wait_asynccnt 0" ::: "memory");
}

// =========================================================================
// Shared prep
// =========================================================================
__global__ void k_zero(float* __restrict__ d, int n) {
  int i = blockIdx.x * blockDim.x + threadIdx.x;
  if (i < n) d[i] = 0.f;
}

__global__ void k_deginv(float* __restrict__ deg, int n) {
  int i = blockIdx.x * blockDim.x + threadIdx.x;
  if (i < n) deg[i] = 1.0f / fmaxf(deg[i], 1.0f);
}

// =========================================================================
// PATH A (large workspace): transposed bf16 operands + async staging
// =========================================================================

// adj [src][dst] fp32 -> maskT [dst][src] bf16 (0/1), fused in-degree count.
__global__ __launch_bounds__(256) void k_maskT_deg(const float* __restrict__ adj,
                                                   unsigned short* __restrict__ maskT,
                                                   float* __restrict__ deg) {
  __shared__ unsigned short lds[64][72];
  const int t = threadIdx.x;
  const long long dstBase = (long long)blockIdx.x * 64;
  const long long srcBase = (long long)blockIdx.y * 64;
  {
    int r = t >> 2, c0 = (t & 3) * 16;
    const float4* p = (const float4*)(adj + (srcBase + r) * NN + dstBase + c0);
#pragma unroll
    for (int q = 0; q < 4; q++) {
      float4 f = p[q];
      lds[r][c0 + q * 4 + 0] = (f.x != 0.f) ? 0x3F80 : 0;
      lds[r][c0 + q * 4 + 1] = (f.y != 0.f) ? 0x3F80 : 0;
      lds[r][c0 + q * 4 + 2] = (f.z != 0.f) ? 0x3F80 : 0;
      lds[r][c0 + q * 4 + 3] = (f.w != 0.f) ? 0x3F80 : 0;
    }
  }
  __syncthreads();
  {
    int d = t >> 2, s0 = (t & 3) * 16;
    union { uint4 q[2]; unsigned short s[16]; } P;
    int cnt = 0;
#pragma unroll
    for (int i = 0; i < 16; i++) {
      unsigned short m = lds[s0 + i][d];
      P.s[i] = m;
      cnt += (m != 0);
    }
    uint4* dp = (uint4*)(maskT + (dstBase + d) * NN + srcBase + s0);
    dp[0] = P.q[0]; dp[1] = P.q[1];
    if (cnt) atomicAdd(&deg[dstBase + d], (float)cnt);  // small-int adds: deterministic
  }
}

// src fp32 [R][C] -> dst bf16 [C][R] (RNE). R,C multiples of 64.
__global__ __launch_bounds__(256) void k_tcvtT(const float* __restrict__ src,
                                               unsigned short* __restrict__ dst,
                                               int R, int C) {
  __shared__ unsigned short lds[64][72];
  const int t = threadIdx.x;
  const long long cBase = (long long)blockIdx.x * 64;
  const long long rBase = (long long)blockIdx.y * 64;
  {
    int r = t >> 2, c0 = (t & 3) * 16;
    const float4* p = (const float4*)(src + (rBase + r) * C + cBase + c0);
#pragma unroll
    for (int q = 0; q < 4; q++) {
      float4 f = p[q];
      lds[r][c0 + q * 4 + 0] = f2bf(f.x);
      lds[r][c0 + q * 4 + 1] = f2bf(f.y);
      lds[r][c0 + q * 4 + 2] = f2bf(f.z);
      lds[r][c0 + q * 4 + 3] = f2bf(f.w);
    }
  }
  __syncthreads();
  {
    int c = t >> 2, r0 = (t & 3) * 16;
    union { uint4 q[2]; unsigned short s[16]; } P;
#pragma unroll
    for (int i = 0; i < 16; i++) P.s[i] = lds[r0 + i][c];
    uint4* dp = (uint4*)(dst + (cBase + c) * R + rBase + r0);
    dp[0] = P.q[0]; dp[1] = P.q[1];
  }
}

// aggT[f][dst] = (sum_src maskT[dst][src] * xT[f][src]) * deg_inv[dst].
// Block: 128 dst x 256 feat. Async double-buffered staging of A (maskT rows)
// and B (xT rows); 16 WMMAs per wave per K-step.
__global__ __launch_bounds__(256) void k_aggT(const unsigned short* __restrict__ maskT,
                                              const unsigned short* __restrict__ xT,
                                              const float* __restrict__ deginv,
                                              unsigned short* __restrict__ aggT) {
  __shared__ unsigned short ldsA[2][128][40];  // [dstLocal][k], 80B row stride
  __shared__ unsigned short ldsB[2][DF][40];   // [feat][k]

  const int t = threadIdx.x;
  const int lane = t & 31, wave = t >> 5;
  const long long dstBase = (long long)blockIdx.x * 128;

  f32x8 acc[16];
#pragma unroll
  for (int i = 0; i < 16; i++)
#pragma unroll
    for (int j = 0; j < 8; j++) acc[i][j] = 0.f;

  auto stage = [&](int buf, int k0) {
#pragma unroll
    for (int j = 0; j < 2; j++) {            // A: 128 rows x 4 chunks = 512
      int ca = t + j * 256;
      int row = ca >> 2, c = (ca & 3) * 8;
      asyncCopy16(&ldsA[buf][row][c], maskT + (dstBase + row) * NN + k0 + c);
    }
#pragma unroll
    for (int j = 0; j < 4; j++) {            // B: 256 rows x 4 chunks = 1024
      int cb = t + j * 256;
      int row = cb >> 2, c = (cb & 3) * 8;
      asyncCopy16(&ldsB[buf][row][c], xT + (long long)row * NN + k0 + c);
    }
  };

  stage(0, 0);
  for (int k0 = 0; k0 < NN; k0 += 32) {
    const int buf = (k0 >> 5) & 1;
    waitAsync0();            // this wave's copies for `buf` have landed
    __syncthreads();         // all waves' copies landed; prior compute done
    if (k0 + 32 < NN) stage(buf ^ 1, k0 + 32);   // overlap with compute

    Frag a;
    const int mRow = wave * 16 + (lane & 15);
    const int kA = (lane < 16) ? 0 : 16;     // A: K kA..kA+15 contiguous
    a.u[0] = *(const uint4*)&ldsA[buf][mRow][kA];
    a.u[1] = *(const uint4*)&ldsA[buf][mRow][kA + 8];
    const int kB = (lane < 16) ? 0 : 16;     // B: K kB..kB+15 contiguous
#pragma unroll
    for (int nt = 0; nt < 16; nt++) {
      Frag b;
      const int nRow = nt * 16 + (lane & 15);
      b.u[0] = *(const uint4*)&ldsB[buf][nRow][kB];
      b.u[1] = *(const uint4*)&ldsB[buf][nRow][kB + 8];
      acc[nt] = __builtin_amdgcn_wmma_f32_16x16x32_bf16(
          false, a.v, false, b.v, (short)0, acc[nt], false, false);
    }
  }

  // Epilogue: lane's 8 accum rows are contiguous nodes -> one b128 per tile.
  const long long rowStart = dstBase + wave * 16 + ((lane < 16) ? 0 : 8);
  float dscale[8];
#pragma unroll
  for (int v = 0; v < 8; v++) dscale[v] = deginv[rowStart + v];
#pragma unroll
  for (int nt = 0; nt < 16; nt++) {
    int col = nt * 16 + (lane & 15);
    union { uint4 q; unsigned short s[8]; } P;
#pragma unroll
    for (int v = 0; v < 8; v++) P.s[v] = f2bf(acc[nt][v] * dscale[v]);
    *(uint4*)(aggT + (long long)col * NN + rowStart) = P.q;
  }
}

// out = x @ Ws + agg @ Wn + b, activations feature-major in/out.
template <int NT, bool F32OUT>
__global__ __launch_bounds__(256) void k_layerT(const unsigned short* __restrict__ xT,
                                                const unsigned short* __restrict__ gT,
                                                const unsigned short* __restrict__ wsT,
                                                const unsigned short* __restrict__ wnT,
                                                const float* __restrict__ bias,
                                                float* __restrict__ outF,
                                                unsigned short* __restrict__ outT) {
  __shared__ unsigned short ldsX[128][40];
  __shared__ unsigned short ldsG[128][40];
  __shared__ unsigned short ldsWs[NT * 16][40];
  __shared__ unsigned short ldsWn[NT * 16][40];

  const int t = threadIdx.x, lane = t & 31, wave = t >> 5;
  const long long rowBase = (long long)blockIdx.x * 128;
  const int Dout = NT * 16;

  f32x8 acc[NT];
#pragma unroll
  for (int i = 0; i < NT; i++)
#pragma unroll
    for (int j = 0; j < 8; j++) acc[i][j] = 0.f;

  const int rp = t >> 4, sub = t & 15, r = rp * 2;

  for (int k0 = 0; k0 < DF; k0 += 32) {
    { // A tiles: transpose feat-major activations -> [node][k]
      const int n0 = sub * 8;
      union { uint4 q; unsigned short s[8]; } X0, X1, G0, G1;
      X0.q = *(const uint4*)(xT + (long long)(k0 + r) * NN + rowBase + n0);
      X1.q = *(const uint4*)(xT + (long long)(k0 + r + 1) * NN + rowBase + n0);
      G0.q = *(const uint4*)(gT + (long long)(k0 + r) * NN + rowBase + n0);
      G1.q = *(const uint4*)(gT + (long long)(k0 + r + 1) * NN + rowBase + n0);
#pragma unroll
      for (int i = 0; i < 8; i++) {
        *(unsigned int*)&ldsX[n0 + i][r] =
            (unsigned int)X0.s[i] | ((unsigned int)X1.s[i] << 16);
        *(unsigned int*)&ldsG[n0 + i][r] =
            (unsigned int)G0.s[i] | ((unsigned int)G1.s[i] << 16);
      }
    }
    // B tiles: pre-transposed weights, K-contiguous rows
    for (int idx = t; idx < NT * 16 * 4; idx += 256) {
      int n = idx >> 2, c = (idx & 3) * 8;
      *(uint4*)&ldsWs[n][c] = *(const uint4*)(wsT + (long long)n * DF + k0 + c);
      *(uint4*)&ldsWn[n][c] = *(const uint4*)(wnT + (long long)n * DF + k0 + c);
    }
    __syncthreads();

    Frag ax, ag;
    const int mRow = wave * 16 + (lane & 15);
    const int kA = (lane < 16) ? 0 : 16;
    ax.u[0] = *(const uint4*)&ldsX[mRow][kA];
    ax.u[1] = *(const uint4*)&ldsX[mRow][kA + 8];
    ag.u[0] = *(const uint4*)&ldsG[mRow][kA];
    ag.u[1] = *(const uint4*)&ldsG[mRow][kA + 8];
    const int kB = (lane < 16) ? 0 : 16;
#pragma unroll
    for (int nt = 0; nt < NT; nt++) {
      Frag bs, bn;
      const int nRow = nt * 16 + (lane & 15);
      bs.u[0] = *(const uint4*)&ldsWs[nRow][kB];
      bs.u[1] = *(const uint4*)&ldsWs[nRow][kB + 8];
      bn.u[0] = *(const uint4*)&ldsWn[nRow][kB];
      bn.u[1] = *(const uint4*)&ldsWn[nRow][kB + 8];
      acc[nt] = __builtin_amdgcn_wmma_f32_16x16x32_bf16(
          false, ax.v, false, bs.v, (short)0, acc[nt], false, false);
      acc[nt] = __builtin_amdgcn_wmma_f32_16x16x32_bf16(
          false, ag.v, false, bn.v, (short)0, acc[nt], false, false);
    }
    __syncthreads();
  }

  const long long rowStart = rowBase + wave * 16 + ((lane < 16) ? 0 : 8);
#pragma unroll
  for (int nt = 0; nt < NT; nt++) {
    int col = nt * 16 + (lane & 15);
    float bcol = bias[col];
    if (F32OUT) {
#pragma unroll
      for (int v = 0; v < 8; v++)
        outF[(rowStart + v) * Dout + col] = acc[nt][v] + bcol;
    } else {
      union { uint4 q; unsigned short s[8]; } P;
#pragma unroll
      for (int v = 0; v < 8; v++) P.s[v] = f2bf(acc[nt][v] + bcol);
      *(uint4*)(outT + (long long)col * NN + rowStart) = P.q;
    }
  }
}

// =========================================================================
// PATH B (small workspace): round-1 on-the-fly kernels
// =========================================================================
__global__ void k_f32_to_bf16(const float* __restrict__ s,
                              unsigned short* __restrict__ d, int n) {
  int i = blockIdx.x * blockDim.x + threadIdx.x;
  if (i < n) d[i] = f2bf(s[i]);
}

__global__ void k_transpose_w(const float* __restrict__ w,
                              unsigned short* __restrict__ wt, int K, int Nc) {
  int i = blockIdx.x * blockDim.x + threadIdx.x;
  if (i >= K * Nc) return;
  int k = i / Nc, n = i % Nc;
  wt[n * K + k] = f2bf(w[i]);
}

__global__ void k_degree(const float* __restrict__ adj, float* __restrict__ deg) {
  long long i = blockIdx.x * (long long)blockDim.x + threadIdx.x;
  const long long stride = (long long)gridDim.x * blockDim.x;
  const long long total = (long long)NN * NN / 4;
  for (; i < total; i += stride) {
    float4 a = ((const float4*)adj)[i];
    int d0 = (int)((i * 4) & (NN - 1));
    if (a.x != 0.f) atomicAdd(&deg[d0 + 0], 1.f);
    if (a.y != 0.f) atomicAdd(&deg[d0 + 1], 1.f);
    if (a.z != 0.f) atomicAdd(&deg[d0 + 2], 1.f);
    if (a.w != 0.f) atomicAdd(&deg[d0 + 3], 1.f);
  }
}

__global__ __launch_bounds__(256) void k_agg(const float* __restrict__ adj,
                                             const unsigned short* __restrict__ xb,
                                             const float* __restrict__ deginv,
                                             unsigned short* __restrict__ agg) {
  __shared__ unsigned short ldsA[128][40];
  __shared__ unsigned short ldsB[DF][40];
  const int t = threadIdx.x;
  const int lane = t & 31, wave = t >> 5;
  const int dstBase = blockIdx.x * 128;
  f32x8 acc[16];
#pragma unroll
  for (int i = 0; i < 16; i++)
#pragma unroll
    for (int j = 0; j < 8; j++) acc[i][j] = 0.f;
  const int sub = t & 15, r = (t >> 4) * 2;
  for (int k0 = 0; k0 < NN; k0 += 32) {
    {
      const int c0 = sub * 8;
      const float* p0 = adj + (long long)(k0 + r) * NN + dstBase + c0;
      const float* p1 = p0 + NN;
      float4 a0 = ((const float4*)p0)[0], a1 = ((const float4*)p0)[1];
      float4 b0 = ((const float4*)p1)[0], b1 = ((const float4*)p1)[1];
      float lo[8] = {a0.x, a0.y, a0.z, a0.w, a1.x, a1.y, a1.z, a1.w};
      float hi[8] = {b0.x, b0.y, b0.z, b0.w, b1.x, b1.y, b1.z, b1.w};
#pragma unroll
      for (int i = 0; i < 8; i++) {
        unsigned int v = (lo[i] != 0.f ? 0x3F80u : 0u) |
                         ((hi[i] != 0.f ? 0x3F80u : 0u) << 16);
        *(unsigned int*)&ldsA[c0 + i][r] = v;
      }
    }
    {
      const int f0 = sub * 16;
      const uint4* q0 = (const uint4*)(xb + (long long)(k0 + r) * DF + f0);
      const uint4* q1 = (const uint4*)(xb + (long long)(k0 + r + 1) * DF + f0);
      union { uint4 u[2]; unsigned short s[16]; } RL, RH;
      RL.u[0] = q0[0]; RL.u[1] = q0[1];
      RH.u[0] = q1[0]; RH.u[1] = q1[1];
#pragma unroll
      for (int i = 0; i < 16; i++) {
        unsigned int v = (unsigned int)RL.s[i] | ((unsigned int)RH.s[i] << 16);
        *(unsigned int*)&ldsB[f0 + i][r] = v;
      }
    }
    __syncthreads();
    Frag a;
    const int mRow = wave * 16 + (lane & 15);
    const int kA = (lane < 16) ? 0 : 16;
    a.u[0] = *(const uint4*)&ldsA[mRow][kA];
    a.u[1] = *(const uint4*)&ldsA[mRow][kA + 8];
    const int kB = (lane < 16) ? 0 : 16;
#pragma unroll
    for (int nt = 0; nt < 16; nt++) {
      Frag b;
      const int nRow = nt * 16 + (lane & 15);
      b.u[0] = *(const uint4*)&ldsB[nRow][kB];
      b.u[1] = *(const uint4*)&ldsB[nRow][kB + 8];
      acc[nt] = __builtin_amdgcn_wmma_f32_16x16x32_bf16(
          false, a.v, false, b.v, (short)0, acc[nt], false, false);
    }
    __syncthreads();
  }
  float dscale[8];
#pragma unroll
  for (int v = 0; v < 8; v++) {
    int row = dstBase + wave * 16 + ((lane < 16) ? v : v + 8);
    dscale[v] = deginv[row];
  }
#pragma unroll
  for (int nt = 0; nt < 16; nt++) {
    int col = nt * 16 + (lane & 15);
#pragma unroll
    for (int v = 0; v < 8; v++) {
      int row = dstBase + wave * 16 + ((lane < 16) ? v : v + 8);
      agg[(long long)row * DF + col] = f2bf(acc[nt][v] * dscale[v]);
    }
  }
}

template <int NT, bool F32OUT>
__global__ __launch_bounds__(256) void k_layer(const unsigned short* __restrict__ x,
                                               const unsigned short* __restrict__ g,
                                               const unsigned short* __restrict__ wsT,
                                               const unsigned short* __restrict__ wnT,
                                               const float* __restrict__ bias,
                                               float* __restrict__ outF,
                                               unsigned short* __restrict__ outB) {
  __shared__ unsigned short ldsX[128][40];
  __shared__ unsigned short ldsG[128][40];
  __shared__ unsigned short ldsWs[NT * 16][40];
  __shared__ unsigned short ldsWn[NT * 16][40];
  const int t = threadIdx.x, lane = t & 31, wave = t >> 5;
  const int rowBase = blockIdx.x * 128;
  const int Dout = NT * 16;
  f32x8 acc[NT];
#pragma unroll
  for (int i = 0; i < NT; i++)
#pragma unroll
    for (int j = 0; j < 8; j++) acc[i][j] = 0.f;
  for (int k0 = 0; k0 < DF; k0 += 32) {
    {
      int m = t >> 1, off = (t & 1) * 16;
      const uint4* px = (const uint4*)(x + (long long)(rowBase + m) * DF + k0 + off);
      const uint4* pg = (const uint4*)(g + (long long)(rowBase + m) * DF + k0 + off);
      *(uint4*)&ldsX[m][off] = px[0];
      *(uint4*)&ldsX[m][off + 8] = px[1];
      *(uint4*)&ldsG[m][off] = pg[0];
      *(uint4*)&ldsG[m][off + 8] = pg[1];
    }
    for (int idx = t; idx < NT * 16 * 4; idx += 256) {
      int n = idx >> 2, c = (idx & 3) * 8;
      *(uint4*)&ldsWs[n][c] = *(const uint4*)(wsT + (long long)n * DF + k0 + c);
      *(uint4*)&ldsWn[n][c] = *(const uint4*)(wnT + (long long)n * DF + k0 + c);
    }
    __syncthreads();
    Frag ax, ag;
    const int mRow = wave * 16 + (lane & 15);
    const int kA = (lane < 16) ? 0 : 16;
    ax.u[0] = *(const uint4*)&ldsX[mRow][kA];
    ax.u[1] = *(const uint4*)&ldsX[mRow][kA + 8];
    ag.u[0] = *(const uint4*)&ldsG[mRow][kA];
    ag.u[1] = *(const uint4*)&ldsG[mRow][kA + 8];
    const int kB = (lane < 16) ? 0 : 16;
#pragma unroll
    for (int nt = 0; nt < NT; nt++) {
      Frag bs, bn;
      const int nRow = nt * 16 + (lane & 15);
      bs.u[0] = *(const uint4*)&ldsWs[nRow][kB];
      bs.u[1] = *(const uint4*)&ldsWs[nRow][kB + 8];
      bn.u[0] = *(const uint4*)&ldsWn[nRow][kB];
      bn.u[1] = *(const uint4*)&ldsWn[nRow][kB + 8];
      acc[nt] = __builtin_amdgcn_wmma_f32_16x16x32_bf16(
          false, ax.v, false, bs.v, (short)0, acc[nt], false, false);
      acc[nt] = __builtin_amdgcn_wmma_f32_16x16x32_bf16(
          false, ag.v, false, bn.v, (short)0, acc[nt], false, false);
    }
    __syncthreads();
  }
#pragma unroll
  for (int nt = 0; nt < NT; nt++) {
    int col = nt * 16 + (lane & 15);
    float bcol = bias[col];
#pragma unroll
    for (int v = 0; v < 8; v++) {
      int row = rowBase + wave * 16 + ((lane < 16) ? v : v + 8);
      float val = acc[nt][v] + bcol;
      if (F32OUT) outF[(long long)row * Dout + col] = val;
      else        outB[(long long)row * Dout + col] = f2bf(val);
    }
  }
}

// =========================================================================
// launch
// =========================================================================
extern "C" void kernel_launch(void* const* d_in, const int* in_sizes, int n_in,
                              void* d_out, int out_size, void* d_ws, size_t ws_size,
                              hipStream_t stream) {
  (void)in_sizes; (void)n_in; (void)out_size;
  const float* h   = (const float*)d_in[0];
  const float* adj = (const float*)d_in[1];
  const float* Ws1 = (const float*)d_in[2];
  const float* Wn1 = (const float*)d_in[3];
  const float* b1  = (const float*)d_in[4];
  const float* Ws2 = (const float*)d_in[5];
  const float* Wn2 = (const float*)d_in[6];
  const float* b2  = (const float*)d_in[7];
  const float* Ws3 = (const float*)d_in[8];
  const float* Wn3 = (const float*)d_in[9];
  const float* b3  = (const float*)d_in[10];

  char* ws = (char*)d_ws;
  const size_t actB = (size_t)NN * DF * 2;                 // 8 MiB
  const size_t maskB = (size_t)NN * NN * 2;                // 512 MiB
  const size_t wB = (size_t)DF * DF * 2, w3B = (size_t)DF * 64 * 2;
  const size_t needA = maskB + 3 * actB + 4 * wB + 2 * w3B + (size_t)NN * 4;

  if (ws_size >= needA) {
    // ---------------- PATH A: maskT + async WMMA pipeline ----------------
    size_t o = 0;
    unsigned short* maskT = (unsigned short*)(ws + o); o += maskB;
    unsigned short* x0T   = (unsigned short*)(ws + o); o += actB;
    unsigned short* x1T   = (unsigned short*)(ws + o); o += actB;
    unsigned short* aggT  = (unsigned short*)(ws + o); o += actB;
    unsigned short* ws1T  = (unsigned short*)(ws + o); o += wB;
    unsigned short* wn1T  = (unsigned short*)(ws + o); o += wB;
    unsigned short* ws2T  = (unsigned short*)(ws + o); o += wB;
    unsigned short* wn2T  = (unsigned short*)(ws + o); o += wB;
    unsigned short* ws3T  = (unsigned short*)(ws + o); o += w3B;
    unsigned short* wn3T  = (unsigned short*)(ws + o); o += w3B;
    float* deg            = (float*)(ws + o);

    k_zero<<<(NN + 255) / 256, 256, 0, stream>>>(deg, NN);
    k_maskT_deg<<<dim3(NN / 64, NN / 64), 256, 0, stream>>>(adj, maskT, deg);
    k_deginv<<<(NN + 255) / 256, 256, 0, stream>>>(deg, NN);
    k_tcvtT<<<dim3(DF / 64, NN / 64), 256, 0, stream>>>(h, x0T, NN, DF);
    k_tcvtT<<<dim3(DF / 64, DF / 64), 256, 0, stream>>>(Ws1, ws1T, DF, DF);
    k_tcvtT<<<dim3(DF / 64, DF / 64), 256, 0, stream>>>(Wn1, wn1T, DF, DF);
    k_tcvtT<<<dim3(DF / 64, DF / 64), 256, 0, stream>>>(Ws2, ws2T, DF, DF);
    k_tcvtT<<<dim3(DF / 64, DF / 64), 256, 0, stream>>>(Wn2, wn2T, DF, DF);
    k_tcvtT<<<dim3(1, DF / 64), 256, 0, stream>>>(Ws3, ws3T, DF, 64);
    k_tcvtT<<<dim3(1, DF / 64), 256, 0, stream>>>(Wn3, wn3T, DF, 64);

    k_aggT<<<NN / 128, 256, 0, stream>>>(maskT, x0T, deg, aggT);
    k_layerT<16, false><<<NN / 128, 256, 0, stream>>>(x0T, aggT, ws1T, wn1T, b1,
                                                      nullptr, x1T);
    k_aggT<<<NN / 128, 256, 0, stream>>>(maskT, x1T, deg, aggT);
    k_layerT<16, false><<<NN / 128, 256, 0, stream>>>(x1T, aggT, ws2T, wn2T, b2,
                                                      nullptr, x0T);
    k_aggT<<<NN / 128, 256, 0, stream>>>(maskT, x0T, deg, aggT);
    k_layerT<4, true><<<NN / 128, 256, 0, stream>>>(x0T, aggT, ws3T, wn3T, b3,
                                                    (float*)d_out, nullptr);
  } else {
    // ---------------- PATH B: on-the-fly (round-1) ----------------
    size_t o = 0;
    unsigned short* x0   = (unsigned short*)(ws + o); o += actB;
    unsigned short* x1   = (unsigned short*)(ws + o); o += actB;
    unsigned short* ag   = (unsigned short*)(ws + o); o += actB;
    unsigned short* ws1T = (unsigned short*)(ws + o); o += wB;
    unsigned short* wn1T = (unsigned short*)(ws + o); o += wB;
    unsigned short* ws2T = (unsigned short*)(ws + o); o += wB;
    unsigned short* wn2T = (unsigned short*)(ws + o); o += wB;
    unsigned short* ws3T = (unsigned short*)(ws + o); o += w3B;
    unsigned short* wn3T = (unsigned short*)(ws + o); o += w3B;
    float* deg           = (float*)(ws + o);

    k_f32_to_bf16<<<(NN * DF + 255) / 256, 256, 0, stream>>>(h, x0, NN * DF);
    k_transpose_w<<<(DF * DF + 255) / 256, 256, 0, stream>>>(Ws1, ws1T, DF, DF);
    k_transpose_w<<<(DF * DF + 255) / 256, 256, 0, stream>>>(Wn1, wn1T, DF, DF);
    k_transpose_w<<<(DF * DF + 255) / 256, 256, 0, stream>>>(Ws2, ws2T, DF, DF);
    k_transpose_w<<<(DF * DF + 255) / 256, 256, 0, stream>>>(Wn2, wn2T, DF, DF);
    k_transpose_w<<<(DF * 64 + 255) / 256, 256, 0, stream>>>(Ws3, ws3T, DF, 64);
    k_transpose_w<<<(DF * 64 + 255) / 256, 256, 0, stream>>>(Wn3, wn3T, DF, 64);
    k_zero<<<(NN + 255) / 256, 256, 0, stream>>>(deg, NN);
    k_degree<<<4096, 256, 0, stream>>>(adj, deg);
    k_deginv<<<(NN + 255) / 256, 256, 0, stream>>>(deg, NN);

    k_agg<<<NN / 128, 256, 0, stream>>>(adj, x0, deg, ag);
    k_layer<16, false><<<NN / 128, 256, 0, stream>>>(x0, ag, ws1T, wn1T, b1,
                                                     nullptr, x1);
    k_agg<<<NN / 128, 256, 0, stream>>>(adj, x1, deg, ag);
    k_layer<16, false><<<NN / 128, 256, 0, stream>>>(x1, ag, ws2T, wn2T, b2,
                                                     nullptr, x0);
    k_agg<<<NN / 128, 256, 0, stream>>>(adj, x0, deg, ag);
    k_layer<4, true><<<NN / 128, 256, 0, stream>>>(x0, ag, ws3T, wn3T, b3,
                                                   (float*)d_out, nullptr);
  }
}
